// Tno_77558519431584
// MI455X (gfx1250) — matrix operand order repaired and checked by reading, not compiled, not run on hardware
//
#include <hip/hip_runtime.h>

// ---------------- problem constants ----------------
#define B_   4
#define H_   8
#define D_   64
#define N_   4096
#define RPE_ 64
#define HD_  (H_*D_)      // 512
#define NLAY 3
#define EPS_ 1e-5f
#define LOG_GAMMA (-1.0005003335835335e-3f)   // ln(0.999)

typedef __attribute__((ext_vector_type(2)))  float    v2f;
typedef __attribute__((ext_vector_type(8)))  float    v8f;
typedef __attribute__((ext_vector_type(16))) _Float16 v16h;
typedef __attribute__((ext_vector_type(8)))  _Float16 v8h;
typedef __attribute__((ext_vector_type(2)))  _Float16 h2;
typedef __attribute__((ext_vector_type(4)))  unsigned int v4u;
typedef __attribute__((ext_vector_type(8)))  int      v8i;
typedef __attribute__((ext_vector_type(4)))  int      v4i;

// =====================================================================
// Kernel 1: RPE MLP, fp32 WMMA 16x16x4.
// One wave (32 threads) per 16-position tile; 256 blocks cover 4096 pos.
// Produces g[m, 512] = rpe(m) (pre-decay, pre-silu) in workspace.
// =====================================================================
__global__ __launch_bounds__(32) void rpe_mlp_kernel(
    const float* __restrict__ pp_w, const float* __restrict__ pp_b,
    const float* __restrict__ ln_w, const float* __restrict__ ln_b,
    const float* __restrict__ lw,   const float* __restrict__ lb,
    const float* __restrict__ oln_w,const float* __restrict__ oln_b,
    const float* __restrict__ out_w,const float* __restrict__ out_b,
    float* __restrict__ g)
{
    __shared__ float actA[16 * 66];
    __shared__ float actB[16 * 66];

    const int lane = threadIdx.x;     // 0..31
    const int row  = lane & 15;       // M row / N col (lane&15)
    const int half = lane >> 4;       // 0 or 1 (lane half)
    const int blk  = blockIdx.x;      // 0..255

    // act[r][j] = pos_r * pp_w[j] + pp_b[j],  pos_r = 16*blk + r
    {
        const float p = (float)(16 * blk + row);
        for (int j = half * 32; j < half * 32 + 32; ++j)
            actA[row * 66 + j] = p * pp_w[j] + pp_b[j];
    }
    __syncthreads();

    // ---- 3 hidden layers: LN -> ReLU -> [16x64]@[64x64] + bias ----
    for (int L = 0; L < NLAY; ++L) {
        if (half == 0) {  // lanes 0-15: one full row each
            float mu = 0.f;
            for (int j = 0; j < 64; ++j) mu += actA[row * 66 + j];
            mu *= (1.f / 64.f);
            float var = 0.f;
            for (int j = 0; j < 64; ++j) {
                float t = actA[row * 66 + j] - mu; var += t * t;
            }
            var *= (1.f / 64.f);
            const float rs = rsqrtf(var + EPS_);
            for (int j = 0; j < 64; ++j) {
                float v = (actA[row * 66 + j] - mu) * rs * ln_w[L * 64 + j]
                          + ln_b[L * 64 + j];
                actB[row * 66 + j] = v > 0.f ? v : 0.f;
            }
        }
        __syncthreads();

        const float* W = lw + L * 64 * 64;   // [k][n] row-major
        v8f acc[4];
        #pragma unroll
        for (int t = 0; t < 4; ++t)
            acc[t] = (v8f){0.f,0.f,0.f,0.f,0.f,0.f,0.f,0.f};

        for (int k = 0; k < 16; ++k) {
            const int kk = 4 * k + 2 * half;      // f32 A: V0=K0|K2, V1=K1|K3
            v2f A;
            A.x = actB[row * 66 + kk];
            A.y = actB[row * 66 + kk + 1];
            #pragma unroll
            for (int t = 0; t < 4; ++t) {
                const int col = 16 * t + row;     // B: row striped across lanes
                v2f Bf;
                Bf.x = W[kk * 64 + col];
                Bf.y = W[(kk + 1) * 64 + col];
                acc[t] = __builtin_amdgcn_wmma_f32_16x16x4_f32(
                    false, A, false, Bf, (short)0, acc[t], false, false);
            }
        }
        __syncthreads();
        #pragma unroll
        for (int t = 0; t < 4; ++t)
            #pragma unroll
            for (int v = 0; v < 8; ++v) {
                const int r2 = v + 8 * half;      // C/D: VGPR v -> M = v or v+8
                const int c2 = 16 * t + row;
                actA[r2 * 66 + c2] = acc[t][v] + lb[L * 64 + c2];
            }
        __syncthreads();
    }

    // ---- output head: LN -> ReLU -> [16x64]@[64x512] + bias -> g ----
    if (half == 0) {
        float mu = 0.f;
        for (int j = 0; j < 64; ++j) mu += actA[row * 66 + j];
        mu *= (1.f / 64.f);
        float var = 0.f;
        for (int j = 0; j < 64; ++j) {
            float t = actA[row * 66 + j] - mu; var += t * t;
        }
        var *= (1.f / 64.f);
        const float rs = rsqrtf(var + EPS_);
        for (int j = 0; j < 64; ++j) {
            float v = (actA[row * 66 + j] - mu) * rs * oln_w[j] + oln_b[j];
            actB[row * 66 + j] = v > 0.f ? v : 0.f;
        }
    }
    __syncthreads();

    for (int t = 0; t < 32; ++t) {                 // 32 N-tiles of 16 -> 512
        v8f acc = (v8f){0.f,0.f,0.f,0.f,0.f,0.f,0.f,0.f};
        for (int k = 0; k < 16; ++k) {
            const int kk = 4 * k + 2 * half;
            v2f A;
            A.x = actB[row * 66 + kk];
            A.y = actB[row * 66 + kk + 1];
            v2f Bf;
            const int col = 16 * t + row;
            Bf.x = out_w[kk * HD_ + col];
            Bf.y = out_w[(kk + 1) * HD_ + col];
            acc = __builtin_amdgcn_wmma_f32_16x16x4_f32(
                false, A, false, Bf, (short)0, acc, false, false);
        }
        #pragma unroll
        for (int v = 0; v < 8; ++v) {
            const int r2 = v + 8 * half;
            const int c2 = 16 * t + row;
            g[(size_t)(16 * blk + r2) * HD_ + c2] = acc[v] + out_b[c2];
        }
    }
}

// =====================================================================
// Kernel 2: causal Toeplitz conv via f16 WMMA 16x16x32.
// One workgroup (8 wave32) per channel (h,d); 512 blocks.
// Each wave owns 8 output tiles (8 x v8f accumulators) and iterates over
// Toeplitz diagonals: one B fragment (a-block, depends only on delta) is
// reused by the 4 same-parity tiles => ~3 ds per wmma instead of ~6.
// The g-column (stride-2KB gather) is DMA'd into LDS by the TDM.
// =====================================================================
__global__ __launch_bounds__(256) void tno_conv_kernel(
    const float* __restrict__ x,    // [B,H,N,D]
    const float* __restrict__ g,    // [N, 512]
    float* __restrict__ out)        // [B,H,N,D]
{
    __shared__ __align__(16) _Float16 Xlds[B_ * N_];  // 32 KB f16 sequences
    __shared__ __align__(16) float    gstage[N_];     // 16 KB TDM destination
    __shared__ _Float16 arev[4224];                   // reversed kernel + guard
    __shared__ h2       apair[4160];                  // (arev[u], arev[u+1])
    __shared__ __align__(16) _Float16 zstub[32];      // zero rows for batch pad

    const int ch  = blockIdx.x;       // 0..511
    const int h   = ch >> 6;
    const int d   = ch & 63;
    const int tid = threadIdx.x;

    if (tid < 32) zstub[tid] = (_Float16)0.f;

    // ---- TDM: DMA g[:, ch] (4096 x f32, stride 512 floats) into gstage ----
    if (tid < 32) {                   // wave 0 only (uniform branch)
        const unsigned long long ga = (unsigned long long)(const void*)(g + ch);
        // LDS aperture truncates flat addr to addr[31:0] -> LDS byte offset
        const unsigned lds_off = (unsigned)(unsigned long long)(void*)gstage;
        v4u g0;
        g0[0] = 1u;                                       // count=1, load, user
        g0[1] = lds_off;                                  // lds_addr
        g0[2] = (unsigned)(ga & 0xFFFFFFFFull);           // global_addr lo
        g0[3] = (unsigned)((ga >> 32) & 0x1FFFFFFull)     // global_addr hi
              | (2u << 30);                               // type = image
        v8i g1;
        g1[0] = 0x00020000;        // workgroup_mask=0, data_size=2 (4 bytes)
        g1[1] = (int)(512u << 16); // tensor_dim0 = 512
        g1[2] = (int)(4096u << 16);// tensor_dim1 = 4096
        g1[3] = (int)(1u << 16);   // tile_dim0 = 1
        g1[4] = 0x00001000;        // tile_dim1 = 4096, tile_dim2 = 0
        g1[5] = 512;               // tensor_dim0_stride = 512 elements
        g1[6] = 0;
        g1[7] = 0;
        __builtin_amdgcn_tensor_load_to_lds(g0, g1, (v4i)0, (v4i)0, (v8i)0, 0);
        __builtin_amdgcn_s_wait_tensorcnt(0);
    }

    // ---- stage x[:, h, :, d] -> f16 (lines L2-resident across channels) ----
    for (int idx = tid; idx < B_ * N_; idx += 256) {
        const int b = idx >> 12;
        const int m = idx & (N_ - 1);
        Xlds[idx] = (_Float16)x[(((size_t)b * H_ + h) * N_ + m) * D_ + d];
    }
    __syncthreads();

    // arev[u] = a[4095-u] (u<=4095) else 0;  a[t] = silu(gamma^t * g[t][ch])
    for (int u = tid; u < 4224; u += 256) {
        float val = 0.f;
        const int t = 4095 - u;
        if (t >= 0) {
            const float gv  = gstage[t];
            const float dec = __expf(LOG_GAMMA * (float)t);
            const float z   = dec * gv;
            val = z / (1.f + __expf(-z));
        }
        arev[u] = (_Float16)val;
    }
    __syncthreads();
    for (int u = tid; u < 4160; u += 256) {
        h2 p; p.x = arev[u]; p.y = arev[u + 1];
        apair[u] = p;
    }
    __syncthreads();

    const int lane  = tid & 31;
    const int wave  = tid >> 5;        // 0..7
    const int j     = lane & 15;       // A: batch row | B/D: column
    const int kk    = lane >> 4;       // lane half
    const int koff1 = 16 * kk;         // f16 A: lane half kk holds K 16kk..16kk+15

#define DO_TILE(t_) do {                                                      \
        const int I_ = i0 + (t_);                                             \
        if (I_ >= dl16) {                                                     \
            const int m0_ = 16 * (I_ - dl16);   /* = 32*J2 */                 \
            const v8h* p1_ = (j < B_)                                         \
                ? (const v8h*)&Xlds[j * N_ + m0_ + koff1]                     \
                : (const v8h*)zstub;                                          \
            const v8h* p2_ = (j < B_)                                         \
                ? (const v8h*)&Xlds[j * N_ + m0_ + koff1 + 8]                 \
                : (const v8h*)zstub;                                          \
            const v8h lo_ = *p1_;                                             \
            const v8h hi_ = *p2_;                                             \
            v16h A_;                                                          \
            _Pragma("unroll")                                                 \
            for (int i_ = 0; i_ < 8; ++i_) {                                  \
                A_[i_] = lo_[i_]; A_[i_ + 8] = hi_[i_];                       \
            }                                                                 \
            acc[t_] = __builtin_amdgcn_wmma_f32_16x16x32_f16(                 \
                false, A_, false, Bf, (short)0, acc[t_], false, false);       \
        }                                                                     \
    } while (0)

    for (int c = wave; c < 32; c += 8) {            // chunk of 8 output tiles
        const int i0 = 8 * c;                       // tiles I = i0 .. i0+7
        v8f acc[8];
        #pragma unroll
        for (int t = 0; t < 8; ++t)
            acc[t] = (v8f){0.f,0.f,0.f,0.f,0.f,0.f,0.f,0.f};

        for (int dl16 = 0; dl16 <= i0 + 7; ++dl16) {    // delta = 16*dl16
            // ---- B fragment: Toeplitz block for this diagonal (shared) ----
            const int u0 = 4095 - 16 * dl16 - j + 16 * kk;
            v16h Bf;
            #pragma unroll
            for (int v = 0; v < 8; ++v) {
                const h2 p = apair[u0 + 2 * v];     // (a[t], a[t-1])
                Bf[2 * v]     = p.x;
                Bf[2 * v + 1] = p.y;
            }
            if ((dl16 & 1) == 0) {                  // even tiles share Bf
                DO_TILE(0); DO_TILE(2); DO_TILE(4); DO_TILE(6);
            } else {                                // odd tiles share Bf
                DO_TILE(1); DO_TILE(3); DO_TILE(5); DO_TILE(7);
            }
        }

        // D layout: lanes 0-15, VGPR v -> M=v (batch), N = j ; only b<4 real
        if (kk == 0) {
            #pragma unroll
            for (int t = 0; t < 8; ++t) {
                const int n = 16 * (i0 + t) + j;
                #pragma unroll
                for (int v = 0; v < B_; ++v)
                    out[(((size_t)v * H_ + h) * N_ + n) * D_ + d] = acc[t][v];
            }
        }
    }
#undef DO_TILE
}

// =====================================================================
extern "C" void kernel_launch(void* const* d_in, const int* in_sizes, int n_in,
                              void* d_out, int out_size, void* d_ws, size_t ws_size,
                              hipStream_t stream) {
    const float* x     = (const float*)d_in[0];
    const float* pp_w  = (const float*)d_in[1];
    const float* pp_b  = (const float*)d_in[2];
    const float* ln_w  = (const float*)d_in[3];
    const float* ln_b  = (const float*)d_in[4];
    const float* lw    = (const float*)d_in[5];
    const float* lb    = (const float*)d_in[6];
    const float* oln_w = (const float*)d_in[7];
    const float* oln_b = (const float*)d_in[8];
    const float* out_w = (const float*)d_in[9];
    const float* out_b = (const float*)d_in[10];

    float* g = (float*)d_ws;                 // [4096, 512] = 8.4 MB scratch

    rpe_mlp_kernel<<<N_ / 16, 32, 0, stream>>>(
        pp_w, pp_b, ln_w, ln_b, lw, lb, oln_w, oln_b, out_w, out_b, g);

    tno_conv_kernel<<<H_ * D_, 256, 0, stream>>>(x, g, (float*)d_out);
}
